// QWen3MoeDecoderLayer_24618752541338
// MI455X (gfx1250) — compile-verified, matrix operand order checked by reading, not run on hardware
//
#include <hip/hip_runtime.h>
#include <hip/hip_bf16.h>

// ---------------------------------------------------------------------------
// Qwen3-MoE decoder layer for MI455X (gfx1250, wave32, WMMA).
// fp32 operands streamed from HBM (bandwidth floor), converted to bf16 on the
// LDS-fill path, v_wmma_f32_16x16x32_bf16 with fp32 accum. GEMMs are double-
// buffered in LDS with register staging so global loads pipeline past the
// WMMA stage. MoE is a gathered grouped-GEMM over the routed top-8 experts;
// the bf16 activation buffer is staged to LDS with async-to-LDS copies
// (ASYNCcnt) when the toolchain exposes the builtin.
// ---------------------------------------------------------------------------

#define T_ 2048
#define D_ 2048
#define HQ_ 32
#define HKV_ 4
#define HD_ 128
#define G_ 8
#define E_ 64
#define TOPK_ 8
#define F_ 768
#define NSLOTS_ (T_*TOPK_)   // 16384

typedef __attribute__((ext_vector_type(16))) __bf16 bf16x16;
typedef __attribute__((ext_vector_type(8)))  __bf16 bf16x8;
typedef __attribute__((ext_vector_type(8)))  float  f32x8;
typedef __attribute__((ext_vector_type(4)))  float  f32x4;

#if defined(__has_builtin)
#  if __has_builtin(__builtin_amdgcn_global_load_async_to_lds_b128)
#    define HAVE_ASYNC_LDS 1
#  endif
#endif
#ifndef HAVE_ASYNC_LDS
#  define HAVE_ASYNC_LDS 0
#endif

#if HAVE_ASYNC_LDS
// Builtin signature (from hipcc diagnostic): param 1 is
//   '__attribute__((__vector_size__(4 * sizeof(int)))) int __device__ *'
// i.e. a v4i32 pointer in addrspace(1); param 2 is the LDS-side pointer.
typedef int v4i_t __attribute__((vector_size(16)));
typedef __attribute__((address_space(1))) v4i_t g_v4i;
typedef __attribute__((address_space(3))) v4i_t l_v4i;
__device__ __forceinline__ void async_copy16(const void* g, void* l) {
  __builtin_amdgcn_global_load_async_to_lds_b128(
      (g_v4i*)(uintptr_t)g,
      (l_v4i*)(uintptr_t)(unsigned)(uintptr_t)l, 0, 0);
}
#endif

// ---- WMMA helpers ----------------------------------------------------------

__device__ __forceinline__ f32x8 wmma_bf16(bf16x16 a, bf16x16 b, f32x8 c) {
  return __builtin_amdgcn_wmma_f32_16x16x32_bf16(
      false, a, false, b, (short)0, c, false, false);
}

// 16x32 bf16 fragment from LDS (row-major, row stride `ld`, ld % 8 == 0 so the
// two ds_load_b128 stay 16B-aligned). Lane layout per CDNA5 ISA 7.12.2.
__device__ __forceinline__ bf16x16 load_frag(const __bf16* base, int ld) {
  const int lane = threadIdx.x & 31;
  const int m = lane & 15, half = lane >> 4;
  const __bf16* p = base + m * ld + half * 8;
  bf16x8 lo = *(const bf16x8*)(p);
  bf16x8 hi = *(const bf16x8*)(p + 16);
  bf16x16 r;
#pragma unroll
  for (int i = 0; i < 8; ++i) { r[i] = lo[i]; r[i + 8] = hi[i]; }
  return r;
}

__device__ __forceinline__ void zero_acc(f32x8& a) {
#pragma unroll
  for (int i = 0; i < 8; ++i) a[i] = 0.f;
}

// ---- RMSNorm ---------------------------------------------------------------

__global__ __launch_bounds__(256)
void rmsnorm_kernel(const float* __restrict__ x, const float* __restrict__ scale,
                    float* __restrict__ y) {
  __shared__ float red[256];
  const int t = blockIdx.x;
  const float* xr = x + (size_t)t * D_;
  float s = 0.f;
  for (int i = threadIdx.x; i < D_; i += 256) { float v = xr[i]; s += v * v; }
  red[threadIdx.x] = s;
  __syncthreads();
  for (int d = 128; d; d >>= 1) {
    if (threadIdx.x < d) red[threadIdx.x] += red[threadIdx.x + d];
    __syncthreads();
  }
  const float r = rsqrtf(red[0] / (float)D_ + 1e-6f);
  for (int i = threadIdx.x; i < D_; i += 256)
    y[(size_t)t * D_ + i] = xr[i] * r * scale[i];
}

// ---- Generic WMMA GEMM: C[M,N] = A[M,K](f32) * B[K,N](f32) (+ residual) ----
// Block tile 128x64, BK=32, 8 waves (wave tile 32x32). Double-buffered LDS,
// register-staged fills, speculative prefetch two tiles ahead.

template <int EPI>  // 0 = store, 1 = store + residual add
__global__ __launch_bounds__(256)
void gemm_wmma_kernel(const float* __restrict__ A, int lda,
                      const float* __restrict__ B, int ldb,
                      float* __restrict__ C, int ldc,
                      const float* __restrict__ R, int K) {
  __shared__ __align__(16) __bf16 Ash[2][128][40];
  __shared__ __align__(16) __bf16 Bsh[2][64][40];   // transposed: [n][k]
  const int tid = threadIdx.x, wave = tid >> 5, lane = tid & 31;
  const int wm = wave & 3, wn = wave >> 2;
  const int half = lane >> 4, nlane = lane & 15;
  const int bm0 = blockIdx.x * 128, bn0 = blockIdx.y * 64;

  f32x4 ra[4], rb[2];
  auto load_tile = [&](int kk) {
#pragma unroll
    for (int i = 0; i < 4; ++i) {
      const int f = tid + i * 256, row = f >> 3, c4 = f & 7;
      ra[i] = *(const f32x4*)(A + (size_t)(bm0 + row) * lda + kk + c4 * 4);
    }
#pragma unroll
    for (int i = 0; i < 2; ++i) {
      const int f = tid + i * 256, kr = f >> 4, c4 = f & 15;
      rb[i] = *(const f32x4*)(B + (size_t)(kk + kr) * ldb + bn0 + c4 * 4);
    }
  };
  auto store_tile = [&](int buf) {
#pragma unroll
    for (int i = 0; i < 4; ++i) {
      const int f = tid + i * 256, row = f >> 3, c4 = f & 7;
#pragma unroll
      for (int j = 0; j < 4; ++j) Ash[buf][row][c4 * 4 + j] = (__bf16)ra[i][j];
    }
#pragma unroll
    for (int i = 0; i < 2; ++i) {
      const int f = tid + i * 256, kr = f >> 4, c4 = f & 15;
#pragma unroll
      for (int j = 0; j < 4; ++j) Bsh[buf][c4 * 4 + j][kr] = (__bf16)rb[i][j];
    }
  };

  f32x8 acc[2][2];
#pragma unroll
  for (int i = 0; i < 2; ++i)
#pragma unroll
    for (int j = 0; j < 2; ++j) zero_acc(acc[i][j]);

  const int nk = K >> 5;
  load_tile(0);
  store_tile(0);
  __syncthreads();

  for (int kt = 0; kt < nk; ++kt) {
    const int buf = kt & 1;
    if (kt + 1 < nk) load_tile((kt + 1) << 5);
    if (kt + 2 < nk) {
      // speculative prefetch: global_prefetch_b8 pulls the whole line
      const int row = tid >> 3, c4 = tid & 7;
      __builtin_prefetch(A + (size_t)(bm0 + row) * lda + ((kt + 2) << 5) + c4 * 4, 0, 1);
      const int kr = tid >> 4, c42 = tid & 15;
      __builtin_prefetch(B + (size_t)(((kt + 2) << 5) + kr) * ldb + bn0 + c42 * 4, 0, 1);
    }
    bf16x16 a0 = load_frag(&Ash[buf][wm * 32][0], 40);
    bf16x16 a1 = load_frag(&Ash[buf][wm * 32 + 16][0], 40);
    bf16x16 b0 = load_frag(&Bsh[buf][wn * 32][0], 40);
    bf16x16 b1 = load_frag(&Bsh[buf][wn * 32 + 16][0], 40);
    acc[0][0] = wmma_bf16(a0, b0, acc[0][0]);
    acc[0][1] = wmma_bf16(a0, b1, acc[0][1]);
    acc[1][0] = wmma_bf16(a1, b0, acc[1][0]);
    acc[1][1] = wmma_bf16(a1, b1, acc[1][1]);
    if (kt + 1 < nk) store_tile(buf ^ 1);
    __syncthreads();
  }

#pragma unroll
  for (int i = 0; i < 2; ++i)
#pragma unroll
    for (int j = 0; j < 2; ++j)
#pragma unroll
      for (int e = 0; e < 8; ++e) {
        const int row = bm0 + wm * 32 + i * 16 + half * 8 + e;
        const int col = bn0 + wn * 32 + j * 16 + nlane;
        float v = acc[i][j][e];
        if (EPI == 1) v += R[(size_t)row * ldc + col];
        C[(size_t)row * ldc + col] = v;
      }
}

// ---- Per-head QK RMSNorm + RoPE (in place) --------------------------------

__global__ __launch_bounds__(128)
void qknorm_rope_kernel(const int* __restrict__ positions,
                        float* __restrict__ q, float* __restrict__ k,
                        const float* __restrict__ qs, const float* __restrict__ ks) {
  const int t = blockIdx.x, h = blockIdx.y, i = threadIdx.x;
  float* p;
  const float* sc;
  if (h < HQ_) { p = q + (size_t)t * (HQ_ * HD_) + h * HD_; sc = qs; }
  else         { p = k + (size_t)t * (HKV_ * HD_) + (h - HQ_) * HD_; sc = ks; }
  __shared__ float sh[128];
  __shared__ float red[128];
  const float v = p[i];
  red[i] = v * v;
  __syncthreads();
  for (int d = 64; d; d >>= 1) {
    if (i < d) red[i] += red[i + d];
    __syncthreads();
  }
  const float r = rsqrtf(red[0] / (float)HD_ + 1e-6f);
  sh[i] = v * r * sc[i];
  __syncthreads();
  const float pos = (float)positions[t];
  const int ii = (i < 64) ? i : i - 64;
  const float inv = __powf(1.0e6f, -(float)ii / 64.f);
  const float ang = pos * inv;
  const float c = cosf(ang), sn = sinf(ang);
  const float x1 = sh[ii], x2 = sh[ii + 64];
  p[i] = (i < 64) ? (x1 * c - x2 * sn) : (x2 * c + x1 * sn);
}

// ---- Flash attention (causal, GQA) ----------------------------------------

__global__ __launch_bounds__(256)
void flash_attn_kernel(const float* __restrict__ q, const float* __restrict__ k,
                       const float* __restrict__ v, float* __restrict__ attn) {
  __shared__ __align__(16) __bf16 Qs[64][136];
  __shared__ __align__(16) __bf16 Ks[32][136];
  __shared__ __align__(16) __bf16 Vs[128][40];   // transposed: [dim][key]
  __shared__ __align__(16) float  Ss[64][36];
  __shared__ __align__(16) __bf16 Ps[64][40];
  __shared__ float m_sh[64], l_sh[64], al_sh[64];

  const int qt = blockIdx.x, qh = blockIdx.y, kvh = qh >> 3;
  const int tid = threadIdx.x, wave = tid >> 5, lane = tid & 31;
  const int half = lane >> 4, nlane = lane & 15;
  const int q0 = qt * 64;

#pragma unroll
  for (int i = 0; i < 8; ++i) {
    const int f = tid + i * 256, row = f >> 5, c4 = f & 31;
    f32x4 vv = *(const f32x4*)(q + (size_t)(q0 + row) * (HQ_ * HD_) + qh * HD_ + c4 * 4);
#pragma unroll
    for (int j = 0; j < 4; ++j) Qs[row][c4 * 4 + j] = (__bf16)vv[j];
  }
  if (tid < 64) { m_sh[tid] = -3.0e38f; l_sh[tid] = 0.f; }

  const int mt = wave >> 1, ntb = (wave & 1) * 4;
  f32x8 acc[4];
#pragma unroll
  for (int i = 0; i < 4; ++i) zero_acc(acc[i]);
  __syncthreads();

  const int nkb = 2 * qt + 2;
  const float scale = 0.08838834764831845f;  // HD^-0.5
  for (int jb = 0; jb < nkb; ++jb) {
    const int kb0 = jb * 32;
    f32x4 rk[4], rv[4];
#pragma unroll
    for (int i = 0; i < 4; ++i) {
      const int f = tid + i * 256, row = f >> 5, c4 = f & 31;
      rk[i] = *(const f32x4*)(k + (size_t)(kb0 + row) * (HKV_ * HD_) + kvh * HD_ + c4 * 4);
      rv[i] = *(const f32x4*)(v + (size_t)(kb0 + row) * (HKV_ * HD_) + kvh * HD_ + c4 * 4);
    }
#pragma unroll
    for (int i = 0; i < 4; ++i) {
      const int f = tid + i * 256, row = f >> 5, c4 = f & 31;
#pragma unroll
      for (int j = 0; j < 4; ++j) {
        Ks[row][c4 * 4 + j] = (__bf16)rk[i][j];
        Vs[c4 * 4 + j][row] = (__bf16)rv[i][j];
      }
    }
    __syncthreads();

    // S = Q K^T : 64x32, one 16x16 fragment per wave
    {
      const int smt = wave >> 1, snt = wave & 1;
      f32x8 s;
      zero_acc(s);
#pragma unroll
      for (int kkk = 0; kkk < HD_; kkk += 32) {
        bf16x16 a = load_frag(&Qs[smt * 16][kkk], 136);
        bf16x16 b = load_frag(&Ks[snt * 16][kkk], 136);
        s = wmma_bf16(a, b, s);
      }
#pragma unroll
      for (int e = 0; e < 8; ++e) {
        const int rl = smt * 16 + half * 8 + e;
        const int cl = snt * 16 + nlane;
        float val = s[e] * scale;
        if (kb0 + cl > q0 + rl) val = -3.0e38f;
        Ss[rl][cl] = val;
      }
    }
    __syncthreads();

    if (tid < 64) {
      const int r = tid;
      const float mold = m_sh[r];
      float mx = mold;
#pragma unroll 8
      for (int c = 0; c < 32; ++c) mx = fmaxf(mx, Ss[r][c]);
      const float alpha = __expf(mold - mx);
      float sum = 0.f;
#pragma unroll 8
      for (int c = 0; c < 32; ++c) {
        const float pv = __expf(Ss[r][c] - mx);
        sum += pv;
        Ps[r][c] = (__bf16)pv;
      }
      m_sh[r] = mx;
      l_sh[r] = l_sh[r] * alpha + sum;
      al_sh[r] = alpha;
    }
    __syncthreads();

    {
      bf16x16 a = load_frag(&Ps[mt * 16][0], 40);
#pragma unroll
      for (int n4 = 0; n4 < 4; ++n4) {
#pragma unroll
        for (int e = 0; e < 8; ++e) acc[n4][e] *= al_sh[mt * 16 + half * 8 + e];
        bf16x16 b = load_frag(&Vs[(ntb + n4) * 16][0], 40);
        acc[n4] = wmma_bf16(a, b, acc[n4]);
      }
    }
    __syncthreads();
  }

#pragma unroll
  for (int n4 = 0; n4 < 4; ++n4)
#pragma unroll
    for (int e = 0; e < 8; ++e) {
      const int rl = mt * 16 + half * 8 + e;
      const int cl = (ntb + n4) * 16 + nlane;
      attn[(size_t)(q0 + rl) * (HQ_ * HD_) + qh * HD_ + cl] = acc[n4][e] / l_sh[rl];
    }
}

// ---- Router softmax + top-8 (one wave per token) ---------------------------

__global__ __launch_bounds__(256)
void router_topk_kernel(const float* __restrict__ logits,
                        int* __restrict__ tidx, float* __restrict__ tw,
                        int* __restrict__ cnt) {
  const int t = blockIdx.x * 8 + (threadIdx.x >> 5);
  const int lane = threadIdx.x & 31;
  const float* lg = logits + (size_t)t * E_;
  const float a0 = lg[lane], a1 = lg[lane + 32];
  float mx = fmaxf(a0, a1);
  for (int d = 16; d; d >>= 1) mx = fmaxf(mx, __shfl_xor(mx, d, 32));
  const float e0 = __expf(a0 - mx), e1 = __expf(a1 - mx);
  float s = e0 + e1;
  for (int d = 16; d; d >>= 1) s += __shfl_xor(s, d, 32);
  float p0 = e0 / s, p1 = e1 / s;

  float myv = 0.f;
  int myi = 0;
  float ssum = 0.f;
  for (int it = 0; it < TOPK_; ++it) {
    float bv;
    int bi;
    if (p0 >= p1) { bv = p0; bi = lane; } else { bv = p1; bi = lane + 32; }
    for (int d = 16; d; d >>= 1) {
      const float ov = __shfl_xor(bv, d, 32);
      const int oi = __shfl_xor(bi, d, 32);
      if (ov > bv || (ov == bv && oi < bi)) { bv = ov; bi = oi; }
    }
    ssum += bv;
    if (lane == it) { myv = bv; myi = bi; }
    if (bi == lane) p0 = -1.f;
    if (bi == lane + 32) p1 = -1.f;
  }
  if (lane < TOPK_) {
    tidx[t * TOPK_ + lane] = myi;
    tw[t * TOPK_ + lane] = myv / ssum;
    atomicAdd(&cnt[myi], 1);
  }
}

__global__ void scan_kernel(const int* __restrict__ cnt, int* __restrict__ offs,
                            int* __restrict__ cur) {
  if (threadIdx.x == 0) {
    int a = 0;
    for (int e = 0; e < E_; ++e) { offs[e] = a; a += cnt[e]; }
  }
  if (threadIdx.x < E_) cur[threadIdx.x] = 0;
}

__global__ __launch_bounds__(256)
void scatter_kernel(const int* __restrict__ tidx, const float* __restrict__ tw,
                    const int* __restrict__ offs, int* __restrict__ cur,
                    int* __restrict__ rows, float* __restrict__ roww) {
  const int g = blockIdx.x * 256 + threadIdx.x;
  const int t = g >> 3;
  const int e = tidx[g];
  const float w = tw[g];
  const int slot = atomicAdd(&cur[e], 1);
  rows[offs[e] + slot] = t;
  roww[offs[e] + slot] = w;
}

// ---- MoE gate/up fused GEMM: act = silu(X@Wg) * (X@Wu), gathered rows ------

__global__ __launch_bounds__(256)
void moe_gateup_kernel(const float* __restrict__ X,
                       const float* __restrict__ WG, const float* __restrict__ WU,
                       const int* __restrict__ cnt, const int* __restrict__ offs,
                       const int* __restrict__ rows, __bf16* __restrict__ act) {
  __shared__ __align__(16) __bf16 Ash[2][128][40];
  __shared__ __align__(16) __bf16 Bg[2][64][40];
  __shared__ __align__(16) __bf16 Bu[2][64][40];
  __shared__ int rid[128];
  const int e = blockIdx.x, f0 = blockIdx.y * 64;
  const int tid = threadIdx.x, wave = tid >> 5, lane = tid & 31;
  const int wm = wave & 3, wn = wave >> 2, half = lane >> 4, nlane = lane & 15;
  const int n = cnt[e], off = offs[e];
  const float* wg = WG + (size_t)e * D_ * F_;
  const float* wu = WU + (size_t)e * D_ * F_;
  const int ntiles = (n + 127) >> 7;
  const int nk = D_ >> 5;

  for (int rt = 0; rt < ntiles; ++rt) {
    if (tid < 128) {
      const int s = rt * 128 + tid;
      rid[tid] = (s < n) ? rows[off + s] : 0;
    }
    __syncthreads();

    f32x4 ra[4], rg[2], ru[2];
    auto load_tile = [&](int kk) {
#pragma unroll
      for (int i = 0; i < 4; ++i) {
        const int f = tid + i * 256, row = f >> 3, c4 = f & 7;
        ra[i] = *(const f32x4*)(X + (size_t)rid[row] * D_ + kk + c4 * 4);
      }
#pragma unroll
      for (int i = 0; i < 2; ++i) {
        const int f = tid + i * 256, kr = f >> 4, c4 = f & 15;
        rg[i] = *(const f32x4*)(wg + (size_t)(kk + kr) * F_ + f0 + c4 * 4);
        ru[i] = *(const f32x4*)(wu + (size_t)(kk + kr) * F_ + f0 + c4 * 4);
      }
    };
    auto store_tile = [&](int buf) {
#pragma unroll
      for (int i = 0; i < 4; ++i) {
        const int f = tid + i * 256, row = f >> 3, c4 = f & 7;
#pragma unroll
        for (int j = 0; j < 4; ++j) Ash[buf][row][c4 * 4 + j] = (__bf16)ra[i][j];
      }
#pragma unroll
      for (int i = 0; i < 2; ++i) {
        const int f = tid + i * 256, kr = f >> 4, c4 = f & 15;
#pragma unroll
        for (int j = 0; j < 4; ++j) {
          Bg[buf][c4 * 4 + j][kr] = (__bf16)rg[i][j];
          Bu[buf][c4 * 4 + j][kr] = (__bf16)ru[i][j];
        }
      }
    };

    f32x8 ag[2][2], au[2][2];
#pragma unroll
    for (int i = 0; i < 2; ++i)
#pragma unroll
      for (int j = 0; j < 2; ++j) { zero_acc(ag[i][j]); zero_acc(au[i][j]); }

    load_tile(0);
    store_tile(0);
    __syncthreads();
    for (int kt = 0; kt < nk; ++kt) {
      const int buf = kt & 1;
      if (kt + 1 < nk) load_tile((kt + 1) << 5);
      if (kt + 2 < nk) {
        const int kr = tid >> 4, c4 = tid & 15;
        __builtin_prefetch(wg + (size_t)(((kt + 2) << 5) + kr) * F_ + f0 + c4 * 4, 0, 1);
        __builtin_prefetch(wu + (size_t)(((kt + 2) << 5) + kr) * F_ + f0 + c4 * 4, 0, 1);
      }
      bf16x16 a0 = load_frag(&Ash[buf][wm * 32][0], 40);
      bf16x16 a1 = load_frag(&Ash[buf][wm * 32 + 16][0], 40);
      bf16x16 g0 = load_frag(&Bg[buf][wn * 32][0], 40);
      bf16x16 g1 = load_frag(&Bg[buf][wn * 32 + 16][0], 40);
      bf16x16 u0 = load_frag(&Bu[buf][wn * 32][0], 40);
      bf16x16 u1 = load_frag(&Bu[buf][wn * 32 + 16][0], 40);
      ag[0][0] = wmma_bf16(a0, g0, ag[0][0]);
      ag[0][1] = wmma_bf16(a0, g1, ag[0][1]);
      ag[1][0] = wmma_bf16(a1, g0, ag[1][0]);
      ag[1][1] = wmma_bf16(a1, g1, ag[1][1]);
      au[0][0] = wmma_bf16(a0, u0, au[0][0]);
      au[0][1] = wmma_bf16(a0, u1, au[0][1]);
      au[1][0] = wmma_bf16(a1, u0, au[1][0]);
      au[1][1] = wmma_bf16(a1, u1, au[1][1]);
      if (kt + 1 < nk) store_tile(buf ^ 1);
      __syncthreads();
    }

#pragma unroll
    for (int i = 0; i < 2; ++i)
#pragma unroll
      for (int j = 0; j < 2; ++j)
#pragma unroll
        for (int ee = 0; ee < 8; ++ee) {
          const int rl = wm * 32 + i * 16 + half * 8 + ee;
          const int slot = rt * 128 + rl;
          if (slot < n) {
            const int cl = wn * 32 + j * 16 + nlane;
            const float g = ag[i][j][ee], u = au[i][j][ee];
            const float hv = (g / (1.f + __expf(-g))) * u;  // silu(g)*u
            act[(size_t)(off + slot) * F_ + f0 + cl] = (__bf16)hv;
          }
        }
  }
}

// ---- MoE down-proj: out[token] += w * (act @ Wd) ---------------------------
// A operand is already bf16 in memory: staged to LDS with async-to-LDS copies
// (ASYNCcnt) when the builtin exists, else register-staged b128 loads.

__global__ __launch_bounds__(256)
void moe_down_kernel(const __bf16* __restrict__ act, const float* __restrict__ WD,
                     const int* __restrict__ cnt, const int* __restrict__ offs,
                     const int* __restrict__ rows, const float* __restrict__ roww,
                     float* __restrict__ out) {
  __shared__ __align__(16) __bf16 Ash[2][128][40];
  __shared__ __align__(16) __bf16 Bs[2][64][40];
  const int e = blockIdx.x, d0 = blockIdx.y * 64;
  const int tid = threadIdx.x, wave = tid >> 5, lane = tid & 31;
  const int wm = wave & 3, wn = wave >> 2, half = lane >> 4, nlane = lane & 15;
  const int n = cnt[e], off = offs[e];
  const float* wd = WD + (size_t)e * F_ * D_;
  const int ntiles = (n + 127) >> 7;
  const int nk = F_ >> 5;

  for (int rt = 0; rt < ntiles; ++rt) {
    f32x4 rb[2];
#if !HAVE_ASYNC_LDS
    bf16x8 raq[2];
#endif
    auto load_a = [&](int kk, int buf) {
#pragma unroll
      for (int i = 0; i < 2; ++i) {
        const int f = tid + i * 256, row = f >> 2, c8 = f & 3;
        const int gs = off + rt * 128 + row;
        if (gs < NSLOTS_) {
#if HAVE_ASYNC_LDS
          async_copy16(act + (size_t)gs * F_ + kk + c8 * 8, &Ash[buf][row][c8 * 8]);
#else
          raq[i] = *(const bf16x8*)(act + (size_t)gs * F_ + kk + c8 * 8);
#endif
        } else {
          bf16x8 z;
#pragma unroll
          for (int j = 0; j < 8; ++j) z[j] = (__bf16)0.f;
#if HAVE_ASYNC_LDS
          *(bf16x8*)&Ash[buf][row][c8 * 8] = z;
#else
          raq[i] = z;
#endif
        }
      }
    };
    auto load_b = [&](int kk) {
#pragma unroll
      for (int i = 0; i < 2; ++i) {
        const int f = tid + i * 256, kr = f >> 4, c4 = f & 15;
        rb[i] = *(const f32x4*)(wd + (size_t)(kk + kr) * D_ + d0 + c4 * 4);
      }
    };
    auto store_tile = [&](int buf) {
#if !HAVE_ASYNC_LDS
#pragma unroll
      for (int i = 0; i < 2; ++i) {
        const int f = tid + i * 256, row = f >> 2, c8 = f & 3;
        *(bf16x8*)&Ash[buf][row][c8 * 8] = raq[i];
      }
#endif
#pragma unroll
      for (int i = 0; i < 2; ++i) {
        const int f = tid + i * 256, kr = f >> 4, c4 = f & 15;
#pragma unroll
        for (int j = 0; j < 4; ++j) Bs[buf][c4 * 4 + j][kr] = (__bf16)rb[i][j];
      }
    };

    f32x8 acc[2][2];
#pragma unroll
    for (int i = 0; i < 2; ++i)
#pragma unroll
      for (int j = 0; j < 2; ++j) zero_acc(acc[i][j]);

    load_a(0, 0);
    load_b(0);
    store_tile(0);
#if HAVE_ASYNC_LDS
    asm volatile("s_wait_asynccnt 0x0" ::: "memory");
#endif
    __syncthreads();

    for (int kt = 0; kt < nk; ++kt) {
      const int buf = kt & 1;
      if (kt + 1 < nk) {
        load_a((kt + 1) << 5, buf ^ 1);   // async copies land in the other buffer
        load_b((kt + 1) << 5);
      }
      if (kt + 2 < nk) {
        const int kr = tid >> 4, c4 = tid & 15;
        __builtin_prefetch(wd + (size_t)(((kt + 2) << 5) + kr) * D_ + d0 + c4 * 4, 0, 1);
      }
      bf16x16 a0 = load_frag(&Ash[buf][wm * 32][0], 40);
      bf16x16 a1 = load_frag(&Ash[buf][wm * 32 + 16][0], 40);
      bf16x16 b0 = load_frag(&Bs[buf][wn * 32][0], 40);
      bf16x16 b1 = load_frag(&Bs[buf][wn * 32 + 16][0], 40);
      acc[0][0] = wmma_bf16(a0, b0, acc[0][0]);
      acc[0][1] = wmma_bf16(a0, b1, acc[0][1]);
      acc[1][0] = wmma_bf16(a1, b0, acc[1][0]);
      acc[1][1] = wmma_bf16(a1, b1, acc[1][1]);
      if (kt + 1 < nk) store_tile(buf ^ 1);
#if HAVE_ASYNC_LDS
      asm volatile("s_wait_asynccnt 0x0" ::: "memory");
#endif
      __syncthreads();
    }

#pragma unroll
    for (int i = 0; i < 2; ++i)
#pragma unroll
      for (int j = 0; j < 2; ++j)
#pragma unroll
        for (int ee = 0; ee < 8; ++ee) {
          const int rl = wm * 32 + i * 16 + half * 8 + ee;
          const int slot = rt * 128 + rl;
          if (slot < n) {
            const int tok = rows[off + slot];
            const float w = roww[off + slot];
            const int cl = wn * 32 + j * 16 + nlane;
            atomicAdd(&out[(size_t)tok * D_ + d0 + cl], w * acc[i][j][ee]);
          }
        }
  }
}

// ---- Host launcher ---------------------------------------------------------

extern "C" void kernel_launch(void* const* d_in, const int* in_sizes, int n_in,
                              void* d_out, int out_size, void* d_ws, size_t ws_size,
                              hipStream_t stream) {
  const int*   positions = (const int*)d_in[0];
  const float* hidden    = (const float*)d_in[1];
  const float* in_ln     = (const float*)d_in[2];
  const float* post_ln   = (const float*)d_in[3];
  const float* q_norm    = (const float*)d_in[4];
  const float* k_norm    = (const float*)d_in[5];
  const float* wq        = (const float*)d_in[6];
  const float* wk        = (const float*)d_in[7];
  const float* wv        = (const float*)d_in[8];
  const float* wo        = (const float*)d_in[9];
  const float* w_router  = (const float*)d_in[10];
  const float* wg        = (const float*)d_in[11];
  const float* wu        = (const float*)d_in[12];
  const float* wd        = (const float*)d_in[13];

  float* out_moe  = (float*)d_out;                       // [T, D]
  float* residual = (float*)d_out + (size_t)T_ * D_;     // [T, D]

  char* ws = (char*)d_ws;
  size_t o = 0;
  auto alloc = [&](size_t bytes) { size_t r = o; o += (bytes + 255) & ~(size_t)255; return r; };
  float* HN   = (float*)(ws + alloc((size_t)T_ * D_ * 4));
  float* Q    = (float*)(ws + alloc((size_t)T_ * HQ_ * HD_ * 4));
  float* K    = (float*)(ws + alloc((size_t)T_ * HKV_ * HD_ * 4));
  float* V    = (float*)(ws + alloc((size_t)T_ * HKV_ * HD_ * 4));
  float* ATT  = (float*)(ws + alloc((size_t)T_ * HQ_ * HD_ * 4));
  float* HN2  = (float*)(ws + alloc((size_t)T_ * D_ * 4));
  float* LOG  = (float*)(ws + alloc((size_t)T_ * E_ * 4));
  int*   TIDX = (int*)  (ws + alloc((size_t)T_ * TOPK_ * 4));
  float* TW   = (float*)(ws + alloc((size_t)T_ * TOPK_ * 4));
  int*   CNT  = (int*)  (ws + alloc(E_ * 4));
  int*   CUR  = (int*)  (ws + alloc(E_ * 4));
  int*   OFFS = (int*)  (ws + alloc(E_ * 4));
  int*   ROWS = (int*)  (ws + alloc((size_t)NSLOTS_ * 4));
  float* ROWW = (float*)(ws + alloc((size_t)NSLOTS_ * 4));
  __bf16* ACT = (__bf16*)(ws + alloc((size_t)NSLOTS_ * F_ * 2));
  (void)ws_size; (void)n_in; (void)in_sizes; (void)out_size;

  (void)hipMemsetAsync(out_moe, 0, (size_t)T_ * D_ * 4, stream);
  (void)hipMemsetAsync(CNT, 0, E_ * 4, stream);

  rmsnorm_kernel<<<T_, 256, 0, stream>>>(hidden, in_ln, HN);

  gemm_wmma_kernel<0><<<dim3(T_ / 128, (HQ_ * HD_) / 64), 256, 0, stream>>>(
      HN, D_, wq, HQ_ * HD_, Q, HQ_ * HD_, nullptr, D_);
  gemm_wmma_kernel<0><<<dim3(T_ / 128, (HKV_ * HD_) / 64), 256, 0, stream>>>(
      HN, D_, wk, HKV_ * HD_, K, HKV_ * HD_, nullptr, D_);
  gemm_wmma_kernel<0><<<dim3(T_ / 128, (HKV_ * HD_) / 64), 256, 0, stream>>>(
      HN, D_, wv, HKV_ * HD_, V, HKV_ * HD_, nullptr, D_);

  qknorm_rope_kernel<<<dim3(T_, HQ_ + HKV_), 128, 0, stream>>>(positions, Q, K, q_norm, k_norm);

  flash_attn_kernel<<<dim3(T_ / 64, HQ_), 256, 0, stream>>>(Q, K, V, ATT);

  gemm_wmma_kernel<1><<<dim3(T_ / 128, D_ / 64), 256, 0, stream>>>(
      ATT, HQ_ * HD_, wo, D_, residual, D_, hidden, HQ_ * HD_);

  rmsnorm_kernel<<<T_, 256, 0, stream>>>(residual, post_ln, HN2);

  gemm_wmma_kernel<0><<<dim3(T_ / 128, 1), 256, 0, stream>>>(
      HN2, D_, w_router, E_, LOG, E_, nullptr, D_);
  router_topk_kernel<<<T_ / 8, 256, 0, stream>>>(LOG, TIDX, TW, CNT);
  scan_kernel<<<1, 64, 0, stream>>>(CNT, OFFS, CUR);
  scatter_kernel<<<NSLOTS_ / 256, 256, 0, stream>>>(TIDX, TW, OFFS, CUR, ROWS, ROWW);

  moe_gateup_kernel<<<dim3(E_, F_ / 64), 256, 0, stream>>>(HN2, wg, wu, CNT, OFFS, ROWS, ACT);
  moe_down_kernel<<<dim3(E_, D_ / 64), 256, 0, stream>>>(ACT, wd, CNT, OFFS, ROWS, ROWW, out_moe);
}
